// Attention_77094662963718
// MI455X (gfx1250) — matrix operand order repaired and checked by reading, not compiled, not run on hardware
//
#include <hip/hip_runtime.h>
#include <hip/hip_bf16.h>

// ---------------- problem constants ----------------
constexpr int Hh   = 8;     // heads
constexpr int SENT = 1024;  // sentence dim
constexpr int HD   = 128;   // per-head hidden dim
constexpr int Nn   = 4096;  // sentences
constexpr int Mm   = 4096;  // knowledge rows
constexpr int KF   = Hh * SENT;  // 8192, K-dim of final GEMM

typedef __bf16 bf16;
typedef __attribute__((ext_vector_type(8)))  __bf16 v8bf;
typedef __attribute__((ext_vector_type(16))) __bf16 v16bf;
typedef __attribute__((ext_vector_type(8)))  float  v8f;
typedef __attribute__((ext_vector_type(4)))  float  v4f;

// ---------------- WMMA helpers ----------------
__device__ __forceinline__ v8f wmma_bf16(v16bf a, v16bf b, v8f c) {
    return __builtin_amdgcn_wmma_f32_16x16x32_bf16(
        /*neg_a=*/false, a, /*neg_b=*/false, b,
        /*c_mod=*/(short)0, c, /*reuse_a=*/false, /*reuse_b=*/false);
}

__device__ __forceinline__ v16bf cat8(v8bf lo, v8bf hi) {
    v16bf r;
#pragma unroll
    for (int i = 0; i < 8; ++i) { r[i] = lo[i]; r[i + 8] = hi[i]; }
    return r;
}

// A operand (16xK row-major bf16): lane holds row (lane&15),
// K = {k0+8*hi .. +7} and {k0+16+8*hi .. +7}  -> two contiguous 16B loads.
__device__ __forceinline__ v16bf load_A_bf16(const bf16* A, int lda, int row0, int k0) {
    int lane = threadIdx.x & 31;
    int r = lane & 15, hi = lane >> 4;
    const bf16* p = A + (long)(row0 + r) * lda + k0 + 8 * hi;
    v8bf a0 = *(const v8bf*)p;
    v8bf a1 = *(const v8bf*)(p + 16);
    return cat8(a0, a1);
}

// Same A layout but source is fp32; convert to bf16 on the fly.
__device__ __forceinline__ v16bf load_A_f32(const float* A, long lda, int row0, int k0) {
    int lane = threadIdx.x & 31;
    int r = lane & 15, hi = lane >> 4;
    const float* p = A + (long)(row0 + r) * lda + k0 + 8 * hi;
    v4f f0 = *(const v4f*)(p);
    v4f f1 = *(const v4f*)(p + 4);
    v4f f2 = *(const v4f*)(p + 16);
    v4f f3 = *(const v4f*)(p + 20);
    v16bf o;
#pragma unroll
    for (int i = 0; i < 4; ++i) {
        o[i]      = (bf16)f0[i];
        o[i + 4]  = (bf16)f1[i];
        o[i + 8]  = (bf16)f2[i];
        o[i + 12] = (bf16)f3[i];
    }
    return o;
}

// B operand from B^T stored row-major [Ncols, K]: lane holds column (lane&15),
// K = {k0+8*hi .. +7} and {k0+16+8*hi .. +7} -> two contiguous 16B loads.
__device__ __forceinline__ v16bf load_B_bf16(const bf16* BT, int ldb, int col0, int k0) {
    int lane = threadIdx.x & 31;
    int c = lane & 15, hi = lane >> 4;
    const bf16* p = BT + (long)(col0 + c) * ldb + k0 + 8 * hi;
    v8bf b0 = *(const v8bf*)p;
    v8bf b1 = *(const v8bf*)(p + 16);
    return cat8(b0, b1);
}

// ---------------- conversion / transpose kernels ----------------
__global__ void k_cvt(const float* __restrict__ in, bf16* __restrict__ out, long n) {
    long i = (long)blockIdx.x * blockDim.x + threadIdx.x;
    if (i < n) out[i] = (bf16)in[i];
}

// in: [batch, R, C] fp32  ->  out: [batch, C, R] bf16   (LDS 32x32 tile)
__global__ void k_transpose_cvt(const float* __restrict__ in, bf16* __restrict__ out,
                                int R, int C) {
    __shared__ float t[32][33];
    long base = (long)blockIdx.z * R * C;
    const float* ip = in + base;
    bf16* op = out + base;
    int c0 = blockIdx.x * 32, r0 = blockIdx.y * 32;
#pragma unroll
    for (int i = threadIdx.y; i < 32; i += 8)
        t[i][threadIdx.x] = ip[(long)(r0 + i) * C + c0 + threadIdx.x];
    __syncthreads();
#pragma unroll
    for (int i = threadIdx.y; i < 32; i += 8)
        op[(long)(c0 + i) * R + r0 + threadIdx.x] = (bf16)t[threadIdx.x][i];
}

// ---------------- GEMM kernels ----------------
// S/K projection: O[h, r, d] = sum_k A[r,k]*W^T[h,d,k] + bias[h,d]   (bf16 out)
// grid (R/128, HD/16, H), block (32,8); one wave per 16x16 tile.
__global__ void k_proj(const bf16* __restrict__ A, const bf16* __restrict__ BT,
                       const float* __restrict__ bias, bf16* __restrict__ O, int R) {
    int h = blockIdx.z;
    int row0 = (blockIdx.x * 8 + threadIdx.y) * 16;
    int col0 = blockIdx.y * 16;
    const bf16* Bh = BT + (long)h * HD * SENT;
    v8f acc = {};
#pragma unroll 4
    for (int k = 0; k < SENT; k += 32) {
        v16bf a = load_A_bf16(A, SENT, row0, k);
        v16bf b = load_B_bf16(Bh, SENT, col0, k);
        acc = wmma_bf16(a, b, acc);
    }
    int lane = threadIdx.x & 31, hi = lane >> 4, c = lane & 15;
    float bv = bias[h * HD + col0 + c];
    bf16* Oh = O + (long)h * R * HD;
#pragma unroll
    for (int i = 0; i < 8; ++i)
        Oh[(long)(row0 + i + 8 * hi) * HD + col0 + c] = (bf16)(acc[i] + bv);
}

// scores[h,n,m] = (S[h,n,:] . K[h,m,:]) / sqrt(HD)   (fp32, into d_out weights region)
// grid (N/128, M/64, H), block (32,8); 16x64 tile per wave (A regs reused x4).
__global__ void k_scores(const bf16* __restrict__ S, const bf16* __restrict__ Kp,
                         float* __restrict__ sc) {
    int h = blockIdx.z;
    int row0 = (blockIdx.x * 8 + threadIdx.y) * 16;
    int col0 = blockIdx.y * 64;
    const bf16* Sh = S + (long)h * Nn * HD;
    const bf16* Kh = Kp + (long)h * Mm * HD;
    v16bf a[4];
#pragma unroll
    for (int k = 0; k < 4; ++k) a[k] = load_A_bf16(Sh, HD, row0, k * 32);
    v8f acc[4] = {};
#pragma unroll
    for (int c = 0; c < 4; ++c)
#pragma unroll
        for (int k = 0; k < 4; ++k) {
            v16bf b = load_B_bf16(Kh, HD, col0 + c * 16, k * 32);
            acc[c] = wmma_bf16(a[k], b, acc[c]);
        }
    const float scale = 0.08838834764831845f;  // 1/sqrt(128)
    int lane = threadIdx.x & 31, hi = lane >> 4, cc = lane & 15;
    float* out = sc + (long)h * Nn * Mm;
#pragma unroll
    for (int c = 0; c < 4; ++c)
#pragma unroll
        for (int i = 0; i < 8; ++i)
            out[(long)(row0 + i + 8 * hi) * Mm + col0 + c * 16 + cc] = acc[c][i] * scale;
}

// Column-wise (over n) online max + sum of exp. grid (M/256, H), block 256.
__global__ void k_stats(const float* __restrict__ sc, float* __restrict__ mx,
                        float* __restrict__ rs) {
    int h = blockIdx.y;
    int m = blockIdx.x * 256 + threadIdx.x;
    const float* p = sc + (long)h * Nn * Mm + m;
    float best = -3.0e38f, s = 0.0f;
    for (int n = 0; n < Nn; ++n) {
        float v = p[(long)n * Mm];
        if (v > best) { s = s * __expf(best - v) + 1.0f; best = v; }
        else          { s += __expf(v - best); }
    }
    mx[h * Mm + m] = best;
    rs[h * Mm + m] = 1.0f / s;
}

// In-place normalize: w = exp(s - mx[h,m]) * rs[h,m].
__global__ void k_norm(float* __restrict__ sc, const float* __restrict__ mx,
                       const float* __restrict__ rs) {
    long idx = (long)blockIdx.x * blockDim.x + threadIdx.x;  // h*N*M + n*M + m
    int m = (int)(idx & (Mm - 1));
    long t = idx >> 12;          // / M
    int h = (int)(t >> 12);      // / N
    float v = sc[idx];
    sc[idx] = __expf(v - mx[h * Mm + m]) * rs[h * Mm + m];
}

// attn: concat[n, h*SENT + d] = sum_m W[h,n,m] * knowledge[m,d]   (bf16 out)
// A = fp32 weights (from d_out) converted on load; B = know^T bf16 [SENT, M].
// grid (N/128, SENT/64, H), block (32,8); 16x64 tile per wave.
__global__ void k_attn(const float* __restrict__ W, const bf16* __restrict__ knowT,
                       bf16* __restrict__ concat) {
    int h = blockIdx.z;
    int row0 = (blockIdx.x * 8 + threadIdx.y) * 16;
    int col0 = blockIdx.y * 64;
    const float* Wh = W + (long)h * Nn * Mm;
    v8f acc[4] = {};
    for (int k = 0; k < Mm; k += 32) {
        v16bf a = load_A_f32(Wh, Mm, row0, k);
#pragma unroll
        for (int c = 0; c < 4; ++c) {
            v16bf b = load_B_bf16(knowT, Mm, col0 + c * 16, k);
            acc[c] = wmma_bf16(a, b, acc[c]);
        }
    }
    int lane = threadIdx.x & 31, hi = lane >> 4, cc = lane & 15;
#pragma unroll
    for (int c = 0; c < 4; ++c)
#pragma unroll
        for (int i = 0; i < 8; ++i)
            concat[(long)(row0 + i + 8 * hi) * KF + h * SENT + col0 + c * 16 + cc] =
                (bf16)acc[c][i];
}

// out[n, o] = sum_k concat[n,k] * W_f[k,o] + b_f[o]   (fp32 out)
// grid (N/128, SENT/64), block (32,8); 16x64 tile per wave.
__global__ void k_final(const bf16* __restrict__ concat, const bf16* __restrict__ WfT,
                        const float* __restrict__ bf, float* __restrict__ out) {
    int row0 = (blockIdx.x * 8 + threadIdx.y) * 16;
    int col0 = blockIdx.y * 64;
    v8f acc[4] = {};
    for (int k = 0; k < KF; k += 32) {
        v16bf a = load_A_bf16(concat, KF, row0, k);
#pragma unroll
        for (int c = 0; c < 4; ++c) {
            v16bf b = load_B_bf16(WfT, KF, col0 + c * 16, k);
            acc[c] = wmma_bf16(a, b, acc[c]);
        }
    }
    int lane = threadIdx.x & 31, hi = lane >> 4, cc = lane & 15;
#pragma unroll
    for (int c = 0; c < 4; ++c) {
        float bias = bf[col0 + c * 16 + cc];
#pragma unroll
        for (int i = 0; i < 8; ++i)
            out[(long)(row0 + i + 8 * hi) * SENT + col0 + c * 16 + cc] = acc[c][i] + bias;
    }
}

// ---------------- launch ----------------
extern "C" void kernel_launch(void* const* d_in, const int* in_sizes, int n_in,
                              void* d_out, int out_size, void* d_ws, size_t ws_size,
                              hipStream_t stream) {
    (void)in_sizes; (void)n_in; (void)out_size; (void)ws_size;
    const float* sentences = (const float*)d_in[0];  // [N, SENT]
    const float* knowledge = (const float*)d_in[1];  // [M, SENT]
    const float* W_s = (const float*)d_in[2];        // [H, SENT, HD]
    const float* b_s = (const float*)d_in[3];        // [H, HD]
    const float* W_k = (const float*)d_in[4];        // [H, SENT, HD]
    const float* b_k = (const float*)d_in[5];        // [H, HD]
    const float* W_f = (const float*)d_in[6];        // [H*SENT, SENT]
    const float* b_f = (const float*)d_in[7];        // [SENT]

    float* out_final = (float*)d_out;                          // [N, SENT]
    float* weights   = out_final + (long)Nn * SENT;            // [H, N, M] = [H*N, M]

    // workspace layout (all sizes already 256B multiples)
    char* w = (char*)d_ws;
    bf16* sent_bf = (bf16*)w;                w += (size_t)Nn * SENT * 2;
    bf16* know_bf = (bf16*)w;                w += (size_t)Mm * SENT * 2;
    bf16* knowT   = (bf16*)w;                w += (size_t)SENT * Mm * 2;
    bf16* WsT     = (bf16*)w;                w += (size_t)Hh * HD * SENT * 2;
    bf16* WkT     = (bf16*)w;                w += (size_t)Hh * HD * SENT * 2;
    bf16* WfT     = (bf16*)w;                w += (size_t)SENT * KF * 2;
    bf16* S_bf    = (bf16*)w;                w += (size_t)Hh * Nn * HD * 2;
    bf16* Kp_bf   = (bf16*)w;                w += (size_t)Hh * Mm * HD * 2;
    bf16* cat_bf  = (bf16*)w;                w += (size_t)Nn * KF * 2;
    float* mx     = (float*)w;               w += (size_t)Hh * Mm * 4;
    float* rs     = (float*)w;

    dim3 wblk(32, 8);

    // 1) converts + transposes
    k_cvt<<<dim3((Nn * SENT) / 256), 256, 0, stream>>>(sentences, sent_bf, (long)Nn * SENT);
    k_cvt<<<dim3((Mm * SENT) / 256), 256, 0, stream>>>(knowledge, know_bf, (long)Mm * SENT);
    k_transpose_cvt<<<dim3(SENT / 32, Mm / 32, 1), wblk, 0, stream>>>(knowledge, knowT, Mm, SENT);
    k_transpose_cvt<<<dim3(HD / 32, SENT / 32, Hh), wblk, 0, stream>>>(W_s, WsT, SENT, HD);
    k_transpose_cvt<<<dim3(HD / 32, SENT / 32, Hh), wblk, 0, stream>>>(W_k, WkT, SENT, HD);
    k_transpose_cvt<<<dim3(SENT / 32, KF / 32, 1), wblk, 0, stream>>>(W_f, WfT, KF, SENT);

    // 2) per-head projections (WMMA)
    k_proj<<<dim3(Nn / 128, HD / 16, Hh), wblk, 0, stream>>>(sent_bf, WsT, b_s, S_bf, Nn);
    k_proj<<<dim3(Mm / 128, HD / 16, Hh), wblk, 0, stream>>>(know_bf, WkT, b_k, Kp_bf, Mm);

    // 3) scores straight into d_out weights region (WMMA)
    k_scores<<<dim3(Nn / 128, Mm / 64, Hh), wblk, 0, stream>>>(S_bf, Kp_bf, weights);

    // 4) column-wise softmax over n, in place
    k_stats<<<dim3(Mm / 256, Hh), 256, 0, stream>>>(weights, mx, rs);
    k_norm<<<dim3((unsigned)(((size_t)Hh * Nn * Mm) / 256)), 256, 0, stream>>>(weights, mx, rs);

    // 5) attention output into concat layout (WMMA, fp32 A converted on load)
    k_attn<<<dim3(Nn / 128, SENT / 64, Hh), wblk, 0, stream>>>(weights, knowT, cat_bf);

    // 6) final projection (WMMA) + bias -> fp32 output
    k_final<<<dim3(Nn / 128, SENT / 64), wblk, 0, stream>>>(cat_bf, WfT, b_f, out_final);
}
